// ResNet_37099927502866
// MI455X (gfx1250) — 8-bit matrix operand order repaired and checked by reading, compiled, not run on hardware
//
#include <hip/hip_runtime.h>
#include <hip/hip_bf16.h>
#include <stdint.h>

#define QSTEP  0.0078125f
#define BN_EPS 1e-5f

typedef __attribute__((ext_vector_type(8))) int          v8i;
typedef __attribute__((ext_vector_type(4))) int          v4i;
typedef __attribute__((ext_vector_type(4))) unsigned int v4u;

#if defined(__has_builtin)
#  if __has_builtin(__builtin_amdgcn_tensor_load_to_lds) && __has_builtin(__builtin_amdgcn_s_wait_tensorcnt)
#    define USE_TDM 1
#  endif
#endif
#ifndef USE_TDM
#  define USE_TDM 0
#endif

static __device__ __forceinline__ float quantize1(float v) {
    float s  = floorf(v * (1.0f / QSTEP));
    float t1 = s * QSTEP;
    float t2 = (s + 1.0f) * QSTEP;
    return (fabsf(v - t1) < fabsf(v - t2)) ? t1 : t2;
}

static __device__ __forceinline__ signed char sgn8(float x) {
    return (x > 0.f) ? (signed char)1 : ((x < 0.f) ? (signed char)-1 : (signed char)0);
}

#if USE_TDM
static __device__ __forceinline__ void tdm_load_2d(unsigned int lds_off, const void* gaddr,
                                                   unsigned int tile_w, unsigned int tile_h,
                                                   unsigned long long stride)
{
    unsigned long long ga = (unsigned long long)(uintptr_t)gaddr;
    v4u g0;
    g0.x = 1u;
    g0.y = lds_off;
    g0.z = (unsigned int)ga;
    g0.w = (unsigned int)((ga >> 32) & 0x1FFFFFFu) | (2u << 30);
    v8i g1;
    g1[0] = 0;
    g1[1] = (int)((tile_w & 0xFFFFu) << 16);
    g1[2] = (int)((tile_w >> 16) | ((tile_h & 0xFFFFu) << 16));
    g1[3] = (int)((tile_h >> 16) | (tile_w << 16));
    g1[4] = (int)(tile_h & 0xFFFFu);
    g1[5] = (int)(unsigned int)stride;
    g1[6] = (int)(unsigned int)((stride >> 32) & 0xFFFFu);
    g1[7] = 0;
    v4i gz = {0, 0, 0, 0};
#if defined(__clang_major__) && __clang_major__ >= 23
    v8i gz8 = {0, 0, 0, 0, 0, 0, 0, 0};
    __builtin_amdgcn_tensor_load_to_lds(g0, g1, gz, gz, gz8, 0);
#else
    __builtin_amdgcn_tensor_load_to_lds(g0, g1, gz, gz, 0);
#endif
}
#endif

__global__ void k_conv1(const float* __restrict__ x,
                        const float* __restrict__ w,
                        float* __restrict__ out,
                        int N, int Co, int Hin, int Win, int Ho, int Wo)
{
    int idx = blockIdx.x * blockDim.x + threadIdx.x;
    int total = N * Co * Ho * Wo;
    if (idx >= total) return;
    int wo = idx % Wo; int t = idx / Wo;
    int ho = t % Ho;   t /= Ho;
    int co = t % Co;   int n = t / Co;
    float acc = 0.f;
    for (int ci = 0; ci < 3; ++ci)
        for (int kh = 0; kh < 7; ++kh) {
            int h = ho * 2 + kh - 3;
            if (h < 0 || h >= Hin) continue;
            for (int kw = 0; kw < 7; ++kw) {
                int ww = wo * 2 + kw - 3;
                if (ww < 0 || ww >= Win) continue;
                float xv = quantize1(x[((n * 3 + ci) * Hin + h) * Win + ww]);
                float wv = w[((co * 3 + ci) * 7 + kh) * 7 + kw];
                acc += xv * (float)sgn8(wv);
            }
        }
    out[idx] = acc;
}

__global__ void k_maxpool(const float* __restrict__ in, float* __restrict__ out,
                          int N, int C, int Hin, int Win, int Ho, int Wo)
{
    int idx = blockIdx.x * blockDim.x + threadIdx.x;
    int total = N * C * Ho * Wo;
    if (idx >= total) return;
    int wo = idx % Wo; int t = idx / Wo;
    int ho = t % Ho;   int nc = t / Ho;
    float mx = -1e30f;
    for (int kh = 0; kh < 3; ++kh) {
        int h = ho * 2 + kh - 1;
        if (h < 0 || h >= Hin) continue;
        for (int kw = 0; kw < 3; ++kw) {
            int w = wo * 2 + kw - 1;
            if (w < 0 || w >= Win) continue;
            mx = fmaxf(mx, in[((size_t)nc * Hin + h) * Win + w]);
        }
    }
    out[idx] = mx;
}

__global__ void k_bn_act(const float* __restrict__ in, float* __restrict__ out,
                         const float* __restrict__ g, const float* __restrict__ b,
                         const float* __restrict__ m, const float* __restrict__ v,
                         int C, int HW, int total, int do_htanh)
{
    int idx = blockIdx.x * blockDim.x + threadIdx.x;
    if (idx >= total) return;
    int c = (idx / HW) % C;
    float y = g[c] * (in[idx] - m[c]) * rsqrtf(v[c] + BN_EPS) + b[c];
    if (do_htanh) y = fminf(1.f, fmaxf(-1.f, y));
    out[idx] = y;
}

__global__ void k_sign_nhwc(const float* __restrict__ in, int8_t* __restrict__ out,
                            int N, int C, int H, int W)
{
    int idx = blockIdx.x * blockDim.x + threadIdx.x;
    int total = N * C * H * W;
    if (idx >= total) return;
    int w = idx % W; int t = idx / W;
    int h = t % H;   t /= H;
    int c = t % C;   int n = t / C;
    out[(((size_t)n * H + h) * W + w) * C + c] = sgn8(in[idx]);
}

__global__ void k_repack_w(const float* __restrict__ w, int8_t* __restrict__ out,
                           int Co, int Ci, int Ci_pad, int KH, int KW,
                           int rows, int K_pad)
{
    int idx = blockIdx.x * blockDim.x + threadIdx.x;
    int total = rows * K_pad;
    if (idx >= total) return;
    int k  = idx % K_pad;
    int co = idx / K_pad;
    int g  = k / Ci_pad;
    int ci = k - g * Ci_pad;
    int8_t s = 0;
    if (co < Co && ci < Ci) {
        int kh = g / KW;
        int kw = g - kh * KW;
        s = sgn8(w[((size_t)(co * Ci + ci) * KH + kh) * KW + kw]);
    }
    out[idx] = s;
}

__global__ void __launch_bounds__(256)
k_bingemm(const int8_t* __restrict__ act,
          const int8_t* __restrict__ wq,
          const float*  __restrict__ bias,
          float* __restrict__ out,
          int N, int Ci, int Ci_pad, int Hin, int Win,
          int Co, int Ho, int Wo,
          int KH, int KW, int stride, int pad, int K_pad)
{
    __shared__ __align__(16) int8_t smem[2 * 128 * 64 + 2 * 16 * 16 * 4];
    int8_t (*ldsA)[128][64] = (int8_t (*)[128][64])smem;
    int    (*ldsB)[16][16]  = (int (*)[16][16])(smem + 2 * 128 * 64);

    const int tid  = threadIdx.x;
    const int lane = tid & 31;
    const int wave = tid >> 5;
    const int half = lane >> 4;
    const int l15  = lane & 15;

    const int Mtiles  = (Co + 15) >> 4;
    const int npos    = N * Ho * Wo;
    const int mblocks = (Mtiles + 7) >> 3;
    const int nt = blockIdx.x / mblocks;
    const int mb = blockIdx.x - nt * mblocks;
    const int mt = mb * 8 + wave;

    const int scol = tid & 15;
    const int skd  = tid >> 4;
    const int spos = nt * 16 + scol;
    const bool s_ok = spos < npos;
    int sn = 0, sho = 0, swo = 0;
    if (s_ok) {
        sn = spos / (Ho * Wo);
        int r = spos - sn * (Ho * Wo);
        sho = r / Wo;
        swo = r - sho * Wo;
    }

    const int pos = nt * 16 + l15;
    const bool pos_ok = pos < npos;
    int n = 0, ho = 0, wo = 0;
    if (pos_ok) {
        n = pos / (Ho * Wo);
        int r = pos - n * (Ho * Wo);
        ho = r / Wo;
        wo = r - ho * Wo;
    }

    const int8_t* wbase = wq + (size_t)(mb * 128) * K_pad;
    const int cblocks = Ci_pad >> 6;
    const int kblocks = KH * KW * cblocks;

    int kh_s = 0, kw_s = 0, cb_s = 0;
    auto stageB = [&](int buf) {
        int ci  = (cb_s << 6) + (skd << 2);
        int val = 0;
        if (s_ok && ci < Ci) {
            int h = sho * stride + kh_s - pad;
            int w = swo * stride + kw_s - pad;
            if (h >= 0 && h < Hin && w >= 0 && w < Win)
                val = *(const int*)(act + (((size_t)(sn * Hin + h) * Win + w) * Ci + ci));
        }
        ldsB[buf][scol][skd] = val;
        if (++cb_s == cblocks) { cb_s = 0; if (++kw_s == KW) { kw_s = 0; ++kh_s; } }
    };

    v8i acc = {0, 0, 0, 0, 0, 0, 0, 0};

#if USE_TDM
    if (wave == 0)
        tdm_load_2d( 0u, wbase, 64u, 128u, (unsigned long long)K_pad);
    stageB(0);
    if (wave == 0) __builtin_amdgcn_s_wait_tensorcnt(0);
    __syncthreads();
#else
    stageB(0);
    __syncthreads();
#endif

    for (int kb = 0; kb < kblocks; ++kb) {
        const int buf = kb & 1;

        if (kb + 1 < kblocks) {
#if USE_TDM
            if (wave == 0)
                tdm_load_2d((unsigned int)(((kb + 1) & 1) * 128 * 64),
                            wbase + ((size_t)(kb + 1) << 6), 64u, 128u,
                            (unsigned long long)K_pad);
#endif
            stageB((kb + 1) & 1);
        }

        v8i A;
#if USE_TDM
        const int8_t* ap = &ldsA[buf][(wave << 4) + l15][0];
        #pragma unroll
        for (int v = 0; v < 8; ++v) {
            const int offA = ((v >> 2) << 5) + (half << 4) + ((v & 3) << 2);
            A[v] = *(const int*)(ap + offA);
        }
#else
        const int8_t* ap = wq + (size_t)(mt * 16 + l15) * K_pad + ((size_t)kb << 6);
        #pragma unroll
        for (int v = 0; v < 8; ++v) {
            const int offA = ((v >> 2) << 5) + (half << 4) + ((v & 3) << 2);
            A[v] = *(const int*)(ap + offA);
        }
#endif

        const int* bp = &ldsB[buf][l15][0];
        const int4 b0 = *(const int4*)(bp + (half << 2));
        const int4 b1 = *(const int4*)(bp + 8 + (half << 2));
        v8i B;
        B[0] = b0.x; B[1] = b0.y; B[2] = b0.z; B[3] = b0.w;
        B[4] = b1.x; B[5] = b1.y; B[6] = b1.z; B[7] = b1.w;

        acc = __builtin_amdgcn_wmma_i32_16x16x64_iu8(true, A, true, B, acc, false, false);

#if USE_TDM
        if (wave == 0 && kb + 1 < kblocks) __builtin_amdgcn_s_wait_tensorcnt(0);
#endif
        __syncthreads();
    }

    if (!pos_ok || mt >= Mtiles) return;
    #pragma unroll
    for (int r = 0; r < 8; ++r) {
        const int co = mt * 16 + r + (half << 3);
        if (co < Co) {
            float val = (float)acc[r];
            if (bias) val += bias[co];
            out[(((size_t)n * Co + co) * Ho + ho) * Wo + wo] = val;
        }
    }
}

__global__ void k_add(float* __restrict__ out, const float* __restrict__ res, int total)
{
    int idx = blockIdx.x * blockDim.x + threadIdx.x;
    if (idx < total) out[idx] += res[idx];
}

__global__ void k_avgpool(const float* __restrict__ in, float* __restrict__ out,
                          int NC, int HW)
{
    int idx = blockIdx.x * blockDim.x + threadIdx.x;
    if (idx >= NC) return;
    const float* p = in + (size_t)idx * HW;
    float s = 0.f;
    for (int i = 0; i < HW; ++i) s += p[i];
    out[idx] = s / (float)HW;
}

__global__ void k_bn_sign_vec(const float* __restrict__ in, int8_t* __restrict__ out,
                              const float* __restrict__ g, const float* __restrict__ b,
                              const float* __restrict__ m, const float* __restrict__ v,
                              int N, int C)
{
    int idx = blockIdx.x * blockDim.x + threadIdx.x;
    if (idx >= N * C) return;
    int c = idx % C;
    float y = g[c] * (in[idx] - m[c]) * rsqrtf(v[c] + BN_EPS) + b[c];
    y = fminf(1.f, fmaxf(-1.f, y));
    out[idx] = sgn8(y);
}

__global__ void k_bn_logsoftmax(const float* __restrict__ in, float* __restrict__ out,
                                const float* __restrict__ g, const float* __restrict__ b,
                                const float* __restrict__ m, const float* __restrict__ v,
                                int C)
{
    __shared__ float red[256];
    const int n = blockIdx.x;
    const int tid = threadIdx.x;

    float mx = -1e30f;
    for (int c = tid; c < C; c += 256) {
        float y = g[c] * (in[n * C + c] - m[c]) * rsqrtf(v[c] + BN_EPS) + b[c];
        mx = fmaxf(mx, y);
    }
    red[tid] = mx; __syncthreads();
    for (int s = 128; s > 0; s >>= 1) {
        if (tid < s) red[tid] = fmaxf(red[tid], red[tid + s]);
        __syncthreads();
    }
    mx = red[0]; __syncthreads();

    float sum = 0.f;
    for (int c = tid; c < C; c += 256) {
        float y = g[c] * (in[n * C + c] - m[c]) * rsqrtf(v[c] + BN_EPS) + b[c];
        sum += expf(y - mx);
    }
    red[tid] = sum; __syncthreads();
    for (int s = 128; s > 0; s >>= 1) {
        if (tid < s) red[tid] += red[tid + s];
        __syncthreads();
    }
    const float lse = logf(red[0]) + mx;
    __syncthreads();

    for (int c = tid; c < C; c += 256) {
        float y = g[c] * (in[n * C + c] - m[c]) * rsqrtf(v[c] + BN_EPS) + b[c];
        out[n * C + c] = y - lse;
    }
}

static inline int cdiv(int a, int b) { return (a + b - 1) / b; }

struct BNP { const float *g, *b, *m, *v; };

extern "C" void kernel_launch(void* const* d_in, const int* in_sizes, int n_in,
                              void* d_out, int out_size, void* d_ws, size_t ws_size,
                              hipStream_t stream)
{
    (void)in_sizes; (void)n_in; (void)out_size; (void)ws_size;
    const int N = 32;
    const int CH[4] = {160, 320, 640, 1280};

    int ii = 0;
    auto F = [&]() { return (const float*)d_in[ii++]; };
    auto takeBN = [&]() { BNP r; r.g = F(); r.b = F(); r.m = F(); r.v = F(); return r; };

    const float* x       = F();
    const float* conv1_w = F();
    BNP bn1 = takeBN();

    struct Blk {
        const float* c1; BNP b1; const float* c2;
        const float* dw; BNP dbn; int has_down;
        BNP b2; int has_b2;
    };
    Blk blk[8];
    int inpl = 160;
    for (int L = 0; L < 4; ++L) {
        int c = CH[L];
        int s = (L == 0) ? 1 : 2;
        Blk b0{}; b0.c1 = F(); b0.b1 = takeBN(); b0.c2 = F();
        if (s != 1 || inpl != c) { b0.dw = F(); b0.dbn = takeBN(); b0.has_down = 1; }
        b0.b2 = takeBN(); b0.has_b2 = 1;
        Blk b1{}; b1.c1 = F(); b1.b1 = takeBN(); b1.c2 = F();
        if (L != 3) { b1.b2 = takeBN(); b1.has_b2 = 1; }
        blk[2 * L] = b0; blk[2 * L + 1] = b1;
        inpl = c;
    }
    BNP bn2 = takeBN();
    const float* fc_w = F();
    const float* fc_b = F();
    BNP bn3 = takeBN();

    char* wsp = (char*)d_ws;
    auto alloc = [&](size_t bytes) -> char* {
        char* p = wsp;
        wsp += (bytes + 255) & ~(size_t)255;
        return p;
    };
    float*  conv1out = (float*)alloc((size_t)N * 160 * 56 * 56 * 4);
    size_t  fbytes   = (size_t)N * 160 * 28 * 28 * 4;
    float*  fb0 = (float*)alloc(fbytes);
    float*  fb1 = (float*)alloc(fbytes);
    float*  fb2 = (float*)alloc(fbytes);
    float*  fb3 = (float*)alloc(fbytes);
    int8_t* s8a = (int8_t*)alloc((size_t)N * 28 * 28 * 160);
    int8_t* wq  = (int8_t*)alloc((size_t)1280 * 11520);
    float*  favg = (float*)alloc((size_t)N * 1280 * 4);
    int8_t* s8fc = (int8_t*)alloc((size_t)N * 1280);
    float*  ffc  = (float*)alloc((size_t)N * 1000 * 4);

    auto repack = [&](const float* w, int Co, int Ci, int KH, int KW) {
        int Ci_pad = cdiv(Ci, 64) * 64;
        int K_pad  = KH * KW * Ci_pad;
        int rows   = cdiv(cdiv(Co, 16), 8) * 128;
        int total  = rows * K_pad;
        k_repack_w<<<cdiv(total, 256), 256, 0, stream>>>(w, wq, Co, Ci, Ci_pad,
                                                         KH, KW, rows, K_pad);
    };
    auto bingemm = [&](const int8_t* a, const float* bias, float* o,
                       int Ci, int Hin, int Win, int Co, int Ho, int Wo,
                       int KH, int KW, int stride, int pad) {
        int Ci_pad  = cdiv(Ci, 64) * 64;
        int K_pad   = KH * KW * Ci_pad;
        int mblocks = cdiv(cdiv(Co, 16), 8);
        int Ntiles  = cdiv(N * Ho * Wo, 16);
        k_bingemm<<<Ntiles * mblocks, 256, 0, stream>>>(a, wq, bias, o,
                                                        N, Ci, Ci_pad, Hin, Win,
                                                        Co, Ho, Wo, KH, KW, stride, pad, K_pad);
    };
    auto bnact = [&](float* buf, BNP p, int C, int HW, int do_htanh) {
        int total = N * C * HW;
        k_bn_act<<<cdiv(total, 256), 256, 0, stream>>>(buf, buf, p.g, p.b, p.m, p.v,
                                                       C, HW, total, do_htanh);
    };

    {
        int total = N * 160 * 56 * 56;
        k_conv1<<<cdiv(total, 256), 256, 0, stream>>>(x, conv1_w, conv1out,
                                                      N, 160, 112, 112, 56, 56);
    }
    float* cur = fb0;
    {
        int total = N * 160 * 28 * 28;
        k_maxpool<<<cdiv(total, 256), 256, 0, stream>>>(conv1out, cur, N, 160, 56, 56, 28, 28);
    }
    bnact(cur, bn1, 160, 28 * 28, 1);

    float* tA = fb1; float* tB = fb2; float* tC = fb3;
    int H = 28, Cin = 160;
    for (int bi = 0; bi < 8; ++bi) {
        const int L = bi >> 1;
        const int c = CH[L];
        const int s = ((bi & 1) == 0 && L > 0) ? 2 : 1;
        const int Ho = (H + 2 - 3) / s + 1;
        const Blk& p = blk[bi];

        {
            int total = N * Cin * H * H;
            k_sign_nhwc<<<cdiv(total, 256), 256, 0, stream>>>(cur, s8a, N, Cin, H, H);
        }
        repack(p.c1, c, Cin, 3, 3);
        bingemm(s8a, nullptr, tA, Cin, H, H, c, Ho, Ho, 3, 3, s, 1);
        bnact(tA, p.b1, c, Ho * Ho, 1);

        float* res;
        if (p.has_down) {
            repack(p.dw, c, Cin, 1, 1);
            bingemm(s8a, nullptr, tC, Cin, H, H, c, Ho, Ho, 1, 1, s, 0);
            bnact(tC, p.dbn, c, Ho * Ho, 0);
            res = tC;
        } else {
            res = cur;
        }

        {
            int total = N * c * Ho * Ho;
            k_sign_nhwc<<<cdiv(total, 256), 256, 0, stream>>>(tA, s8a, N, c, Ho, Ho);
        }
        repack(p.c2, c, c, 3, 3);
        bingemm(s8a, nullptr, tB, c, Ho, Ho, c, Ho, Ho, 3, 3, 1, 1);

        {
            int total = N * c * Ho * Ho;
            k_add<<<cdiv(total, 256), 256, 0, stream>>>(tB, res, total);
        }
        if (p.has_b2) bnact(tB, p.b2, c, Ho * Ho, 1);

        float* old = cur; cur = tB; tB = old;
        H = Ho; Cin = c;
    }

    k_avgpool<<<cdiv(N * 1280, 256), 256, 0, stream>>>(cur, favg, N * 1280, H * H);
    k_bn_sign_vec<<<cdiv(N * 1280, 256), 256, 0, stream>>>(favg, s8fc,
                                                           bn2.g, bn2.b, bn2.m, bn2.v, N, 1280);
    repack(fc_w, 1000, 1280, 1, 1);
    bingemm(s8fc, fc_b, ffc, 1280, 1, 1, 1000, 1, 1, 1, 1, 1, 0);
    k_bn_logsoftmax<<<N, 256, 0, stream>>>(ffc, (float*)d_out,
                                           bn3.g, bn3.b, bn3.m, bn3.v, 1000);
}
